// EnsembleConvRNN_85779086836153
// MI455X (gfx1250) — matrix operand order repaired and checked by reading, not compiled, not run on hardware
//
#include <hip/hip_runtime.h>
#include <hip/hip_bf16.h>

// ---------------------------------------------------------------------------
// EnsembleConvRNN for MI455X (gfx1250).
// B=32, L=256, D=H=256, E=8, NL=4, K=4.  All GEMMs via v_wmma_f32_16x16x32_bf16.
// Recurrence keeps W_hh fragments resident in VGPRs (wave32 has 1024 VGPRs),
// hidden state in LDS, and prefetches the next step's `pre` slice.
// ---------------------------------------------------------------------------

typedef __attribute__((ext_vector_type(16))) __bf16 v16bf;
typedef __attribute__((ext_vector_type(8)))  float  v8f;

#define BB 32
#define LL 256
#define DD 256
#define HH 256
#define EE 8
#define NLAYER 4

__device__ __forceinline__ v8f wmma_bf16(v16bf a, v16bf b, v8f c) {
    // (neg_a, A, neg_b, B, c_mod, C, reuse_a, reuse_b)
    return __builtin_amdgcn_wmma_f32_16x16x32_bf16(false, a, false, b, (short)0, c,
                                                   false, false);
}

// ---------------------------------------------------------------------------
// f32 -> bf16 conversion (weights), n multiple of 256.
// ---------------------------------------------------------------------------
__global__ void k_cvt(const float* __restrict__ src, __bf16* __restrict__ dst) {
    size_t i = (size_t)blockIdx.x * 256 + threadIdx.x;
    dst[i] = (__bf16)src[i];
}

// ---------------------------------------------------------------------------
// LayerNorm over D.  One block per (b,l), 256 threads.
// ---------------------------------------------------------------------------
__global__ void k_ln(const float* __restrict__ x, const float* __restrict__ g,
                     const float* __restrict__ bta, float* __restrict__ o) {
    const int bl = blockIdx.x;
    const int d  = threadIdx.x;
    __shared__ float red[256];
    float v = x[(size_t)bl * DD + d];
    red[d] = v;
    __syncthreads();
    for (int s = 128; s > 0; s >>= 1) {
        if (d < s) red[d] += red[d + s];
        __syncthreads();
    }
    float mean = red[0] * (1.0f / DD);
    __syncthreads();
    float c = v - mean;
    red[d] = c * c;
    __syncthreads();
    for (int s = 128; s > 0; s >>= 1) {
        if (d < s) red[d] += red[d + s];
        __syncthreads();
    }
    float var = red[0] * (1.0f / DD);
    o[(size_t)bl * DD + d] = c * rsqrtf(var + 1e-5f) * g[d] + bta[d];
}

// ---------------------------------------------------------------------------
// Depthwise causal conv1d (K=4, left pad 3).  One block per (b,l).
// ---------------------------------------------------------------------------
__global__ void k_conv(const float* __restrict__ ln, const float* __restrict__ cw,
                       const float* __restrict__ cb, float* __restrict__ o) {
    const int bl = blockIdx.x;
    const int b  = bl / LL;
    const int l  = bl % LL;
    const int d  = threadIdx.x;
    float acc = cb[d];
#pragma unroll
    for (int j = 0; j < 4; ++j) {
        int li = l - 3 + j;
        if (li >= 0) acc += ln[((size_t)b * LL + li) * DD + d] * cw[d * 4 + j];
    }
    o[(size_t)bl * DD + d] = acc;
}

// ---------------------------------------------------------------------------
// Expand conv output across ensemble with layer-0 input scaling r0:
// A0[(b,e,l),f] = conv[(b,l),f] * r0[e,f]  (bf16)
// ---------------------------------------------------------------------------
__global__ void k_expand(const float* __restrict__ conv, const float* __restrict__ r0,
                         __bf16* __restrict__ A0) {
    size_t i   = (size_t)blockIdx.x * 256 + threadIdx.x;  // over B*E*L*D
    int    f   = (int)(i & 255);
    size_t bel = i >> 8;
    int    l   = (int)(bel & 255);
    size_t be  = bel >> 8;
    int    e   = (int)(be & 7);
    int    b   = (int)(be >> 3);
    float  v   = conv[(((size_t)b * LL) + l) * DD + f] * r0[e * DD + f];
    A0[i] = (__bf16)v;
}

// ---------------------------------------------------------------------------
// Input projection GEMM: pre[row,n] = (A @ W_ih^T)[row,n] * s[e,n] + bias[n]
// A: (M=B*E*L, 256) bf16 row-major.  W: (256,256) bf16 row-major (B[k,n]=W[n,k]).
// Block = 256 threads (8 waves), 64x64 tile; wave (wm 0..3, wn 0..1) does 16x32.
// ---------------------------------------------------------------------------
__global__ void k_proj(const __bf16* __restrict__ A, const __bf16* __restrict__ W,
                       const float* __restrict__ s, const float* __restrict__ bias,
                       float* __restrict__ pre) {
    const int tid  = threadIdx.x;
    const int w    = tid >> 5;
    const int lane = tid & 31;
    const int half = lane >> 4;
    const int l16  = lane & 15;
    const int wm   = w & 3;
    const int wn   = w >> 2;
    const int row0 = blockIdx.x * 64 + wm * 16;
    const int col0 = blockIdx.y * 64 + wn * 32;
    const int rowA = row0 + l16;

    v8f acc[2] = {};
    for (int kk = 0; kk < 256; kk += 32) {
        // A fragment (16x32): lane holds row l16, 16 contiguous K at half*16
        v16bf a;
        const __bf16* pa = A + (size_t)rowA * 256 + kk;
#pragma unroll
        for (int j = 0; j < 8; ++j) {
            a[j]     = pa[half * 16 + j];
            a[8 + j] = pa[half * 16 + 8 + j];
        }
#pragma unroll
        for (int t = 0; t < 2; ++t) {
            // B fragment (32x16): lane holds col n, 16 contiguous K at half*16
            int n = col0 + t * 16 + l16;
            const __bf16* pw = W + (size_t)n * 256 + kk + half * 16;
            v16bf bb;
#pragma unroll
            for (int j = 0; j < 16; ++j) bb[j] = pw[j];
            acc[t] = wmma_bf16(a, bb, acc[t]);
        }
    }
    // Epilogue: *s[e,n] + bias[n].  C layout: m = r + 8*half, n = l16.
#pragma unroll
    for (int t = 0; t < 2; ++t) {
        int n = col0 + t * 16 + l16;
#pragma unroll
        for (int r2 = 0; r2 < 8; ++r2) {
            int row = row0 + r2 + 8 * half;
            int e   = (row >> 8) & 7;  // row = (b*E+e)*L + l, L=256
            pre[(size_t)row * HH + n] = acc[t][r2] * s[e * HH + n] + bias[n];
        }
    }
}

// ---------------------------------------------------------------------------
// RNN recurrence: each block owns 16 'be' rows (recurrence is independent per
// row).  W_hh fragments for this wave's 32 columns are loaded ONCE into VGPRs
// (16 x v16bf = 128 VGPRs) and reused for all 256 steps; hidden state h
// (16x256 bf16, 8KB) lives in LDS.  Next step's `pre` slice is prefetched.
// ---------------------------------------------------------------------------
__global__ void k_rnn(const float* __restrict__ pre, const __bf16* __restrict__ Whh,
                      __bf16* __restrict__ ys_bf, float* __restrict__ out_f32,
                      float* __restrict__ hlast, int lastLayer) {
    __shared__ __bf16 hl[16 * 256];
    const int tid  = threadIdx.x;
    const int w    = tid >> 5;
    const int lane = tid & 31;
    const int half = lane >> 4;
    const int l16  = lane & 15;
    const int ncol0  = w * 32;
    const int beBase = blockIdx.x * 16;

    // Hoist B fragments (W_hh) into registers: bfrag[t][kk/32]
    v16bf bfrag[2][8];
#pragma unroll
    for (int t = 0; t < 2; ++t) {
        int n = ncol0 + t * 16 + l16;
#pragma unroll
        for (int k8 = 0; k8 < 8; ++k8) {
            const __bf16* pb = Whh + (size_t)n * 256 + k8 * 32 + half * 16;
#pragma unroll
            for (int j = 0; j < 16; ++j) bfrag[t][k8][j] = pb[j];
        }
    }

    for (int i = tid; i < 4096; i += 256) hl[i] = (__bf16)0.0f;
    __syncthreads();

    for (int l = 0; l < LL; ++l) {
        v8f acc[2] = {};
#pragma unroll
        for (int k8 = 0; k8 < 8; ++k8) {
            v16bf a;
            const __bf16* pa = hl + l16 * 256 + k8 * 32;  // A rows = the 16 be rows
#pragma unroll
            for (int j = 0; j < 8; ++j) {
                a[j]     = pa[half * 16 + j];
                a[8 + j] = pa[half * 16 + 8 + j];
            }
            acc[0] = wmma_bf16(a, bfrag[0][k8], acc[0]);
            acc[1] = wmma_bf16(a, bfrag[1][k8], acc[1]);
        }
        // Prefetch next step's pre slice (16 rows x 1KB; 64B per thread).
        if (l + 1 < LL) {
            const float* pf =
                pre + ((size_t)(beBase + (tid >> 4)) * LL + (l + 1)) * HH + (tid & 15) * 16;
            __builtin_prefetch(pf, 0, 1);
        }
        __syncthreads();  // all h reads done before overwrite
#pragma unroll
        for (int t = 0; t < 2; ++t) {
            int n = ncol0 + t * 16 + l16;
#pragma unroll
            for (int r2 = 0; r2 < 8; ++r2) {
                int    m    = r2 + 8 * half;
                int    be   = beBase + m;
                size_t oidx = ((size_t)be * LL + l) * HH + n;
                float  hv   = tanhf(acc[t][r2] + pre[oidx]);
                hl[m * 256 + n] = (__bf16)hv;
                if (lastLayer) {
                    out_f32[oidx] = hv;
                    if (l == LL - 1) hlast[(size_t)be * HH + n] = hv;
                } else {
                    ys_bf[oidx] = (__bf16)hv;
                }
            }
        }
        __syncthreads();  // h_new visible before next step reads
    }
}

// ---------------------------------------------------------------------------
extern "C" void kernel_launch(void* const* d_in, const int* in_sizes, int n_in,
                              void* d_out, int out_size, void* d_ws, size_t ws_size,
                              hipStream_t stream) {
    const float* x      = (const float*)d_in[0];
    const float* conv_w = (const float*)d_in[1];
    const float* conv_b = (const float*)d_in[2];
    const float* ln_g   = (const float*)d_in[3];
    const float* ln_b   = (const float*)d_in[4];
    const float* W_ih   = (const float*)d_in[5];  // (NL,H,D)
    const float* W_hh   = (const float*)d_in[6];  // (NL,H,H)
    const float* r      = (const float*)d_in[7];  // (NL,E,D)
    const float* s      = (const float*)d_in[8];  // (NL,E,H)
    const float* bvec   = (const float*)d_in[9];  // (NL,H)
    float* out = (float*)d_out;

    char* ws = (char*)d_ws;
    float*  ln      = (float*)(ws);                         // 8 MB  (B*L*D f32)
    float*  conv    = (float*)(ws + ((size_t)8 << 20));     // 8 MB
    __bf16* bufA    = (__bf16*)(ws + ((size_t)16 << 20));   // 32 MB (B*E*L*D bf16)
    __bf16* bufB    = (__bf16*)(ws + ((size_t)48 << 20));   // 32 MB
    float*  pre     = (float*)(ws + ((size_t)80 << 20));    // 64 MB (B*E*L*H f32)
    __bf16* Wih_bf  = (__bf16*)(ws + ((size_t)144 << 20));  // 512 KB (NL*H*D bf16)
    __bf16* Whh_bf  = (__bf16*)(ws + ((size_t)145 << 20));  // 512 KB (NL*H*H bf16)

    const int WN = NLAYER * HH * DD;  // 262144
    k_cvt<<<WN / 256, 256, 0, stream>>>(W_ih, Wih_bf);
    k_cvt<<<WN / 256, 256, 0, stream>>>(W_hh, Whh_bf);

    k_ln<<<BB * LL, 256, 0, stream>>>(x, ln_g, ln_b, ln);
    k_conv<<<BB * LL, 256, 0, stream>>>(ln, conv_w, conv_b, conv);
    k_expand<<<(BB * EE * LL * DD) / 256, 256, 0, stream>>>(conv, r, bufA);

    __bf16* Ain = bufA;
    __bf16* Ys  = bufB;
    float*  hlast = out + (size_t)BB * EE * LL * HH;
    for (int i = 0; i < NLAYER; ++i) {
        k_proj<<<dim3((BB * EE * LL) / 64, HH / 64), 256, 0, stream>>>(
            Ain, Wih_bf + (size_t)i * HH * DD, s + (size_t)i * EE * HH,
            bvec + (size_t)i * HH, pre);
        k_rnn<<<(BB * EE) / 16, 256, 0, stream>>>(
            pre, Whh_bf + (size_t)i * HH * HH, Ys, out, hlast,
            i == NLAYER - 1 ? 1 : 0);
        __bf16* tmp = Ain; Ain = Ys; Ys = tmp;
    }
}
